// NonAutoregressiveMultiGNNv1_28183575396637
// MI455X (gfx1250) — matrix operand order repaired and checked by reading, not compiled, not run on hardware
//
#include <hip/hip_runtime.h>
#include <hip/hip_bf16.h>

// ---------------------------------------------------------------------------
// GVP-GNN forward on MI455X (gfx1250).
// Scalar-path matmuls run on v_wmma_f32_16x16x32_bf16 with branch-free,
// software-pipelined inner loops: M%32==0 everywhere, K padded to %32 in
// activations, weights packed to K-padded bf16 once per launch, A staged in
// fragment order in double-buffered LDS (one barrier per k-tile, 2 WMMAs).
// LN / vector einsums / gathers / scatters are VALU (bandwidth-trivial).
// ---------------------------------------------------------------------------

typedef __attribute__((ext_vector_type(16))) __bf16 v16bf;
typedef __attribute__((ext_vector_type(8)))  float  v8f;

#define N_NODES 8000
#define N_CONF  2
#define K_NBR   16
#define N_EDGES (N_NODES * K_NBR)          // 128000
#define NC      (N_NODES * N_CONF)         // 16000
#define EC      (N_EDGES * N_CONF)         // 256000
#define RCH     16384                      // edge-conf rows per chunk (mult of 32)

static inline int ceil_div(int a, int b) { return (a + b - 1) / b; }
static inline int round32(int a) { return (a + 31) & ~31; }

__device__ inline unsigned short bfb(float x) {   // f32 -> bf16 bits (RNE)
  unsigned u = __float_as_uint(x);
  u += 0x7FFFu + ((u >> 16) & 1u);
  return (unsigned short)(u >> 16);
}
__device__ inline __bf16 f2bf(float x) { return __builtin_bit_cast(__bf16, bfb(x)); }
__device__ inline float sigmoidf_(float x) { return 1.f / (1.f + __expf(-x)); }

// ---------------------------------------------------------------------------
// GEMM: Y[M,N] = X[M,Kp] @ Wp[N,Kp]^T + bias.   Preconditions:
//   M % 32 == 0, Kp % 32 == 0, ldx % 4 == 0, X 16B-aligned, Wp bf16 packed.
// Block = 128 threads (4 waves): 32 M rows (2 tiles/wave) x 64 N columns.
// A staged in fragment order in double-buffered LDS: one __syncthreads per
// k-tile, next tile's global loads overlap current tile's WMMAs, and each
// B fragment (one contiguous 32B load of packed bf16) feeds two WMMAs.
// A and B fragments both use contiguous k = 16*half + t.
// ---------------------------------------------------------------------------
__global__ void gemm_bias_bf16wmma(const float* __restrict__ X, int ldx,
                                   const __bf16* __restrict__ Wp,  // [N,Kp]
                                   const float* __restrict__ bias, // [N] or null
                                   float* __restrict__ Y, int ldy,
                                   int M, int N, int Kp) {
  __shared__ __align__(32) unsigned short Afrag[2][2][32 * 16]; // [buf][mtile][lane*16+t]
  const int tid  = threadIdx.x;
  const int wave = tid >> 5, lane = tid & 31;
  const int half = lane >> 4, l16 = lane & 15;
  const int m0 = blockIdx.x * 32;
  const int n0 = blockIdx.y * 64 + wave * 16;
  const int col = n0 + l16;
  const bool colok = col < N;
  const __bf16* wr = Wp + (size_t)(colok ? col : 0) * Kp + (half << 4);

  // staging: thread tid loads A[m0+mm][kk..kk+3] (tile0) and row +16 (tile1);
  // a 4-aligned k group maps affinely into lane (mm + 16*sthalf) elements
  // tbase..tbase+3:  k in [0,16): t=k (half0), [16,32): t=k-16 (half1)
  const int mm = tid >> 3;
  const int kk = (tid & 7) << 2;
  const int sthalf = (kk & 16) ? 1 : 0;
  const int tbase  = kk & 15;
  const int sidx   = ((mm + (sthalf << 4)) << 4) + tbase;
  const float* xr0 = X + (size_t)(m0 + mm) * ldx + kk;
  const float* xr1 = xr0 + (size_t)16 * ldx;

  // prologue: stage k-tile 0 into buffer 0
  float4 xa = *(const float4*)(xr0);
  float4 xb = *(const float4*)(xr1);
  {
    ushort4 p0; p0.x = bfb(xa.x); p0.y = bfb(xa.y); p0.z = bfb(xa.z); p0.w = bfb(xa.w);
    ushort4 p1; p1.x = bfb(xb.x); p1.y = bfb(xb.y); p1.z = bfb(xb.z); p1.w = bfb(xb.w);
    *(ushort4*)&Afrag[0][0][sidx] = p0;
    *(ushort4*)&Afrag[0][1][sidx] = p1;
  }
  __syncthreads();

  v8f acc0 = {}, acc1 = {};
  int p = 0;
  for (int k0 = 0; k0 < Kp; k0 += 32) {
    const v16bf b = *(const v16bf*)(wr + k0);
    int kn = k0 + 32;                       // branch-free prefetch index
    kn = (kn < Kp) ? kn : 0;                // last iter re-stages tile 0 (unused)
    xa = *(const float4*)(xr0 + kn);
    xb = *(const float4*)(xr1 + kn);
    const v16bf a0 = *(const v16bf*)&Afrag[p][0][lane << 4];
    const v16bf a1 = *(const v16bf*)&Afrag[p][1][lane << 4];
    acc0 = __builtin_amdgcn_wmma_f32_16x16x32_bf16(false, a0, false, b,
                                                   (short)0, acc0, false, false);
    acc1 = __builtin_amdgcn_wmma_f32_16x16x32_bf16(false, a1, false, b,
                                                   (short)0, acc1, false, false);
    ushort4 p0; p0.x = bfb(xa.x); p0.y = bfb(xa.y); p0.z = bfb(xa.z); p0.w = bfb(xa.w);
    ushort4 p1; p1.x = bfb(xb.x); p1.y = bfb(xb.y); p1.z = bfb(xb.z); p1.w = bfb(xb.w);
    *(ushort4*)&Afrag[p ^ 1][0][sidx] = p0;
    *(ushort4*)&Afrag[p ^ 1][1][sidx] = p1;
    __syncthreads();
    p ^= 1;
  }

  if (colok) {
    const float bn = bias ? bias[col] : 0.f;
#pragma unroll
    for (int i = 0; i < 8; ++i) {
      int m = m0 + i + (half << 3);
      Y[(size_t)m * ldy + col] = acc0[i] + bn;
      Y[(size_t)(m + 16) * ldy + col] = acc1[i] + bn;
    }
  }
}

// pack fp32 weight [N,K] -> bf16 [N,Kp] with zero K padding
__global__ void pack_w_bf16(const float* __restrict__ W, __bf16* __restrict__ Wp,
                            int N, int K, int Kp) {
  int idx = blockIdx.x * blockDim.x + threadIdx.x;
  if (idx >= N * Kp) return;
  int n = idx / Kp, k = idx % Kp;
  Wp[(size_t)n * Kp + k] = f2bf(k < K ? W[(size_t)n * K + k] : 0.f);
}

// ---------------------------------------------------------------------------
// VALU kernels
// ---------------------------------------------------------------------------
__global__ void fillf(float* p, float v, int n) {
  int i = blockIdx.x * blockDim.x + threadIdx.x;
  if (i < n) p[i] = v;
}

__global__ void count_dst_k(const int* __restrict__ dst, float* cnt, int E) {
  int e = blockIdx.x * blockDim.x + threadIdx.x;
  if (e < E) atomicAdd(cnt + dst[e], 1.f);
}

// s68 = [node_s | 0.25 x4]  (pairing probs are analytically 0.25)
__global__ void build_x68(const float* __restrict__ ns, float* __restrict__ o, int R) {
  int i = blockIdx.x * blockDim.x + threadIdx.x;
  if (i >= R * 68) return;
  int r = i / 68, j = i % 68;
  o[(size_t)r * 68 + j] = (j < 64) ? ns[(size_t)r * 64 + j] : 0.25f;
}

// gvp.LayerNorm: standard LN over S scalars; RMS over V vector channels.
__global__ void ln_kernel(const float* __restrict__ s_in, int ld_in,
                          const float* __restrict__ v_in,
                          const float* __restrict__ w, const float* __restrict__ b,
                          float* __restrict__ s_out, int ld_out,
                          float* __restrict__ v_out, int R, int S, int V) {
  int r = blockIdx.x * blockDim.x + threadIdx.x;
  if (r >= R) return;
  const float* sr = s_in + (size_t)r * ld_in;
  float mu = 0.f;
  for (int j = 0; j < S; ++j) mu += sr[j];
  mu /= S;
  float var = 0.f;
  for (int j = 0; j < S; ++j) { float d = sr[j] - mu; var += d * d; }
  var /= S;
  float rstd = rsqrtf(var + 1e-5f);
  float* so = s_out + (size_t)r * ld_out;
  for (int j = 0; j < S; ++j) so[j] = (sr[j] - mu) * rstd * w[j] + b[j];
  if (V > 0) {
    const float* vr = v_in + (size_t)r * V * 3;
    float sum = 0.f;
    for (int vv = 0; vv < V; ++vv) {
      float x = vr[vv * 3], y = vr[vv * 3 + 1], z = vr[vv * 3 + 2];
      sum += fmaxf(x * x + y * y + z * z, 1e-8f);
    }
    float rs = rsqrtf(sum / V);
    float* vo = v_out + (size_t)r * V * 3;
    for (int t = 0; t < V * 3; ++t) vo[t] = vr[t] * rs;
  }
}

// vh = wh @ v (per row); vn = sqrt(clip(|vh|^2,eps)) appended to Xcat col xoff+o
__global__ void gvp_vec_pre(const float* __restrict__ Vin,  // [R,vi,3]
                            const float* __restrict__ wh,   // [h,vi]
                            float* __restrict__ vh,         // [R,h,3]
                            float* __restrict__ Xcat, int ldx, int xoff,
                            int R, int vi, int h) {
  int idx = blockIdx.x * blockDim.x + threadIdx.x;
  if (idx >= R * h) return;
  int r = idx / h, o = idx % h;
  const float* vr = Vin + (size_t)r * vi * 3;
  const float* wr = wh + (size_t)o * vi;
  float a0 = 0.f, a1 = 0.f, a2 = 0.f;
  for (int v = 0; v < vi; ++v) {
    float wv = wr[v];
    a0 += wv * vr[v * 3]; a1 += wv * vr[v * 3 + 1]; a2 += wv * vr[v * 3 + 2];
  }
  float* out = vh + ((size_t)r * h + o) * 3;
  out[0] = a0; out[1] = a1; out[2] = a2;
  Xcat[(size_t)r * ldx + xoff + o] = sqrtf(fmaxf(a0 * a0 + a1 * a1 + a2 * a2, 1e-8f));
}

// v_out[r,o,:] = (wv @ vh)[o,:] * sigmoid(G[r,o])
__global__ void gate_vec_out(const float* __restrict__ vh, const float* __restrict__ wv,
                             const float* __restrict__ G, float* __restrict__ Vout,
                             int R, int h, int vo) {
  int idx = blockIdx.x * blockDim.x + threadIdx.x;
  if (idx >= R * vo) return;
  int r = idx / vo, o = idx % vo;
  const float* vr = vh + (size_t)r * h * 3;
  const float* wr = wv + (size_t)o * h;
  float a0 = 0.f, a1 = 0.f, a2 = 0.f;
  for (int k = 0; k < h; ++k) {
    float w = wr[k];
    a0 += w * vr[k * 3]; a1 += w * vr[k * 3 + 1]; a2 += w * vr[k * 3 + 2];
  }
  float g = sigmoidf_(G[(size_t)r * vo + o]);
  float* out = Vout + ((size_t)r * vo + o) * 3;
  out[0] = a0 * g; out[1] = a1 * g; out[2] = a2 * g;
}

__global__ void silu_cols(float* __restrict__ X, int ld, int nc, int R) {
  int idx = blockIdx.x * blockDim.x + threadIdx.x;
  if (idx >= R * nc) return;
  int r = idx / nc, c = idx % nc;
  float x = X[(size_t)r * ld + c];
  X[(size_t)r * ld + c] = x * sigmoidf_(x);
}

// X0[:,0:288] = [ns[src] | es | ns[dst]]
__global__ void gather_msg_s(const float* __restrict__ ns, const float* __restrict__ es,
                             const int* __restrict__ src, const int* __restrict__ dst,
                             float* __restrict__ X0, int ldx, int base, int Rcur) {
  int idx = blockIdx.x * blockDim.x + threadIdx.x;
  if (idx >= Rcur * 288) return;
  int r = idx / 288, col = idx % 288;
  int g = base + r, e = g >> 1, c = g & 1;
  float v;
  if (col < 128)      v = ns[((size_t)src[e] * 2 + c) * 128 + col];
  else if (col < 160) v = es[((size_t)e * 2 + c) * 32 + (col - 128)];
  else                v = ns[((size_t)dst[e] * 2 + c) * 128 + (col - 160)];
  X0[(size_t)r * ldx + col] = v;
}

// MV[:,0:33,:] = [nv[src](16) | ev(1) | nv[dst](16)]
__global__ void gather_msg_v(const float* __restrict__ nv, const float* __restrict__ ev,
                             const int* __restrict__ src, const int* __restrict__ dst,
                             float* __restrict__ MV, int base, int Rcur) {
  int idx = blockIdx.x * blockDim.x + threadIdx.x;
  if (idx >= Rcur * 99) return;
  int r = idx / 99, t = idx % 99, vv = t / 3, cc = t % 3;
  int g = base + r, e = g >> 1, c = g & 1;
  float v;
  if (vv < 16)       v = nv[(((size_t)src[e] * 2 + c) * 16 + vv) * 3 + cc];
  else if (vv == 16) v = ev[((size_t)e * 2 + c) * 3 + cc];
  else               v = nv[(((size_t)dst[e] * 2 + c) * 16 + (vv - 17)) * 3 + cc];
  MV[(size_t)r * 99 + t] = v;
}

__global__ void scatter_add(const float* __restrict__ Xsrc, const int* __restrict__ dst,
                            float* __restrict__ acc, int per, int base, int Rcur) {
  int idx = blockIdx.x * blockDim.x + threadIdx.x;
  if (idx >= Rcur * per) return;
  int r = idx / per, j = idx % per;
  int g = base + r, e = g >> 1, c = g & 1;
  atomicAdd(acc + ((size_t)dst[e] * 2 + c) * per + j, Xsrc[(size_t)r * per + j]);
}

__global__ void residual_mean(float* __restrict__ s, const float* __restrict__ acc,
                              const float* __restrict__ cnt, int per, int R) {
  int idx = blockIdx.x * blockDim.x + threadIdx.x;
  if (idx >= R * per) return;
  int rc = idx / per;
  float d = fmaxf(cnt[rc >> 1], 1.f);
  s[idx] += acc[idx] / d;
}

__global__ void add_inplace(float* __restrict__ a, const float* __restrict__ b, int n) {
  int i = blockIdx.x * blockDim.x + threadIdx.x;
  if (i < n) a[i] += b[i];
}

__global__ void pool_conf(const float* __restrict__ x, float* __restrict__ y, int per, int N) {
  int idx = blockIdx.x * blockDim.x + threadIdx.x;
  if (idx >= N * per) return;
  int n = idx / per, j = idx % per;
  y[idx] = 0.5f * (x[((size_t)n * 2) * per + j] + x[((size_t)n * 2 + 1) * per + j]);
}

// ---------------------------------------------------------------------------
// Host-side orchestration
// ---------------------------------------------------------------------------
struct GVPp {
  const float *wh, *ws_b, *ws_w, *wsv_b, *wsv_w, *wv;
  __bf16 *ws_p, *wsv_p;
  int si, vi, so, vo, h, kp;
};
struct LNp { const float *b, *w; };

static void launch_gemm(hipStream_t st, const float* X, int ldx, const __bf16* Wp,
                        const float* bias, float* Y, int ldy, int M, int N, int Kp) {
  dim3 g(M / 32, ceil_div(N, 64));
  gemm_bias_bf16wmma<<<g, dim3(128), 0, st>>>(X, ldx, Wp, bias, Y, ldy, M, N, Kp);
}

static void run_gvp(hipStream_t st, const GVPp& p, int R,
                    float* Xcat, int ldx, const float* Vin, float* vhb,
                    float* Sout, int ldS, float* Gb, float* Vout, bool act) {
  int n = R * p.h;
  gvp_vec_pre<<<ceil_div(n, 256), 256, 0, st>>>(Vin, p.wh, vhb, Xcat, ldx, p.si, R, p.vi, p.h);
  launch_gemm(st, Xcat, ldx, p.ws_p, p.ws_b, Sout, ldS, R, p.so, p.kp);
  if (p.vo > 0) {
    launch_gemm(st, Sout, ldS, p.wsv_p, p.wsv_b, Gb, p.vo, R, p.vo, p.so);
    n = R * p.vo;
    gate_vec_out<<<ceil_div(n, 256), 256, 0, st>>>(vhb, p.wv, Gb, Vout, R, p.h, p.vo);
  }
  if (act) {
    n = R * p.so;
    silu_cols<<<ceil_div(n, 256), 256, 0, st>>>(Sout, ldS, p.so, R);
  }
}

extern "C" void kernel_launch(void* const* d_in, const int* in_sizes, int n_in,
                              void* d_out, int out_size, void* d_ws, size_t ws_size,
                              hipStream_t stream) {
  (void)in_sizes; (void)n_in; (void)out_size; (void)ws_size;
  const float* node_s = (const float*)d_in[0];   // [8000,2,64]
  const float* node_v = (const float*)d_in[1];   // [8000,2,4,3]
  const float* edge_s = (const float*)d_in[2];   // [E,2,32]
  const float* edge_v = (const float*)d_in[3];   // [E,2,1,3]
  const int*   eidx   = (const int*)d_in[4];     // [2,E]
  const int *src = eidx, *dst = eidx + N_EDGES;

  // ---- workspace allocator (floats, 16B-aligned chunks) ----
  float* wsf = (float*)d_ws;
  size_t off = 0;
  auto alloc = [&](size_t n) { float* p = wsf + off; off += (n + 3) & ~(size_t)3; return p; };
  auto allocb = [&](size_t n) { return (__bf16*)alloc((n + 1) / 2); };   // bf16 elems

  auto packW = [&](const float* W, int N, int K, int Kp) {
    __bf16* p = allocb((size_t)N * Kp);
    pack_w_bf16<<<ceil_div(N * Kp, 256), 256, 0, stream>>>(W, p, N, K, Kp);
    return p;
  };

  // ---- params: pytree-flatten order (dict keys sorted, lists in order) ----
  int pi = 5;
  auto nextp = [&]() { return (const float*)d_in[pi++]; };
  auto getGVP = [&](int si, int vi, int so, int vo) {
    GVPp p; p.si = si; p.vi = vi; p.so = so; p.vo = vo; p.h = vo > vi ? vo : vi;
    p.wh = nextp(); p.ws_b = nextp(); p.ws_w = nextp();
    if (vo) { p.wsv_b = nextp(); p.wsv_w = nextp(); p.wv = nextp(); }
    else    { p.wsv_b = p.wsv_w = p.wv = nullptr; }
    p.kp = round32(si + p.h);
    p.ws_p = packW(p.ws_w, so, si + p.h, p.kp);
    p.wsv_p = vo ? packW(p.wsv_w, vo, so, so) : nullptr;   // so is %32 already
    return p;
  };
  auto getLN = [&]() { LNp l; l.b = nextp(); l.w = nextp(); return l; };

  GVPp We_gvp = getGVP(32, 1, 32, 1);
  LNp  We_ln  = getLN();
  GVPp Wv_gvp = getGVP(68, 4, 128, 16);
  LNp  Wv_ln  = getLN();
  struct Layer { GVPp ff0, ff1; LNp ln0, ln1; GVPp msg[3]; } L[3];
  for (int i = 0; i < 3; ++i) {
    L[i].ff0 = getGVP(128, 16, 512, 32);
    L[i].ff1 = getGVP(512, 32, 128, 16);
    L[i].ln0 = getLN();
    L[i].ln1 = getLN();
    L[i].msg[0] = getGVP(288, 33, 128, 16);
    L[i].msg[1] = getGVP(128, 16, 128, 16);
    L[i].msg[2] = getGVP(128, 16, 128, 16);
  }
  GVPp out_gvp1 = getGVP(128, 16, 128, 16);
  GVPp out_gvp2 = getGVP(128, 16, 4, 0);
  LNp  out_ln   = getLN();

  // ---- activation buffers (K-padded leading dims, pad columns zeroed) ----
  float* cnt   = alloc(N_NODES);
  float* s68   = alloc((size_t)NC * 68);
  float* X84   = alloc((size_t)NC * 96);     // ld 96  (K=84)
  float* v0ln  = alloc((size_t)NC * 12);
  float* vh16  = alloc((size_t)NC * 48);
  float* Gn    = alloc((size_t)NC * 16);
  float* s     = alloc((size_t)NC * 128);
  float* v     = alloc((size_t)NC * 48);
  float* Xe    = alloc((size_t)EC * 64);     // ld 64  (K=33)
  float* evln  = alloc((size_t)EC * 3);
  float* vhe   = alloc((size_t)EC * 3);
  float* Ge    = alloc((size_t)EC * 1);
  float* es    = alloc((size_t)EC * 32);
  float* ev    = alloc((size_t)EC * 3);
  float* ns    = alloc((size_t)NC * 128);
  float* nv    = alloc((size_t)NC * 48);
  float* acc_s = alloc((size_t)NC * 128);
  float* acc_v = alloc((size_t)NC * 48);
  float* Xf0   = alloc((size_t)NC * 160);    // ld 160 (K=160, no pad)
  float* Xf1   = alloc((size_t)NC * 576);    // ld 576 (K=544)
  float* vhf0  = alloc((size_t)NC * 96);
  float* vhf1  = alloc((size_t)NC * 96);
  float* Gf    = alloc((size_t)NC * 32);
  float* Vf0   = alloc((size_t)NC * 96);
  float* Sf    = alloc((size_t)NC * 128);
  float* Vf1   = alloc((size_t)NC * 48);
  float* sp    = alloc((size_t)N_NODES * 128);
  float* vp    = alloc((size_t)N_NODES * 48);
  float* vpl   = alloc((size_t)N_NODES * 48);
  float* Xo1   = alloc((size_t)N_NODES * 160);  // ld 160 (K=144)
  float* Xo2   = alloc((size_t)N_NODES * 160);  // ld 160 (K=144)
  float* vho   = alloc((size_t)N_NODES * 48);
  float* vho2  = alloc((size_t)N_NODES * 48);
  float* Go    = alloc((size_t)N_NODES * 16);
  float* Vo    = alloc((size_t)N_NODES * 48);
  // per-chunk buffers
  float* X0  = alloc((size_t)RCH * 352);     // ld 352 (K=321)
  float* MV  = alloc((size_t)RCH * 99);
  float* vh0 = alloc((size_t)RCH * 99);
  float* X1  = alloc((size_t)RCH * 160);     // ld 160 (K=144)
  float* vh1 = alloc((size_t)RCH * 48);
  float* V1  = alloc((size_t)RCH * 48);
  float* X2  = alloc((size_t)RCH * 160);     // ld 160 (K=144)
  float* vh2 = alloc((size_t)RCH * 48);
  float* V2  = alloc((size_t)RCH * 48);
  float* S3  = alloc((size_t)RCH * 128);
  float* V3  = alloc((size_t)RCH * 48);
  float* Gc  = alloc((size_t)RCH * 16);

  auto zero = [&](float* p, int n) { fillf<<<ceil_div(n, 256), 256, 0, stream>>>(p, 0.f, n); };

  // zero K-padded buffers once (pad columns are never rewritten afterwards)
  zero(X84, NC * 96);
  zero(Xe, EC * 64);
  zero(Xf1, NC * 576);
  zero(Xo1, N_NODES * 160);
  zero(Xo2, N_NODES * 160);
  zero(X0, RCH * 352);
  zero(X1, RCH * 160);
  zero(X2, RCH * 160);

  // 1) dst degree for conv mean aggregation
  zero(cnt, N_NODES);
  count_dst_k<<<ceil_div(N_EDGES, 256), 256, 0, stream>>>(dst, cnt, N_EDGES);

  // 2) node embedding: concat(node_s, pp=0.25) -> LN -> GVP (no act)
  build_x68<<<ceil_div(NC * 68, 256), 256, 0, stream>>>(node_s, s68, NC);
  ln_kernel<<<ceil_div(NC, 128), 128, 0, stream>>>(s68, 68, node_v, Wv_ln.w, Wv_ln.b,
                                                   X84, 96, v0ln, NC, 68, 4);
  run_gvp(stream, Wv_gvp, NC, X84, 96, v0ln, vh16, s, 128, Gn, v, false);

  // 3) edge embedding: LN -> GVP (no act)
  ln_kernel<<<ceil_div(EC, 128), 128, 0, stream>>>(edge_s, 32, edge_v, We_ln.w, We_ln.b,
                                                   Xe, 64, evln, EC, 32, 1);
  run_gvp(stream, We_gvp, EC, Xe, 64, evln, vhe, es, 32, Ge, ev, false);

  // 4) 3 conv + FF layers
  for (int li = 0; li < 3; ++li) {
    const Layer& lp = L[li];
    ln_kernel<<<ceil_div(NC, 128), 128, 0, stream>>>(s, 128, v, lp.ln0.w, lp.ln0.b,
                                                     ns, 128, nv, NC, 128, 16);
    zero(acc_s, NC * 128);
    zero(acc_v, NC * 48);
    for (int base = 0; base < EC; base += RCH) {
      int R = (EC - base < RCH) ? (EC - base) : RCH;
      gather_msg_s<<<ceil_div(R * 288, 256), 256, 0, stream>>>(ns, es, src, dst, X0, 352, base, R);
      gather_msg_v<<<ceil_div(R * 99, 256), 256, 0, stream>>>(nv, ev, src, dst, MV, base, R);
      run_gvp(stream, lp.msg[0], R, X0, 352, MV, vh0, X1, 160, Gc, V1, true);
      run_gvp(stream, lp.msg[1], R, X1, 160, V1, vh1, X2, 160, Gc, V2, true);
      run_gvp(stream, lp.msg[2], R, X2, 160, V2, vh2, S3, 128, Gc, V3, false);
      scatter_add<<<ceil_div(R * 128, 256), 256, 0, stream>>>(S3, dst, acc_s, 128, base, R);
      scatter_add<<<ceil_div(R * 48, 256), 256, 0, stream>>>(V3, dst, acc_v, 48, base, R);
    }
    residual_mean<<<ceil_div(NC * 128, 256), 256, 0, stream>>>(s, acc_s, cnt, 128, NC);
    residual_mean<<<ceil_div(NC * 48, 256), 256, 0, stream>>>(v, acc_v, cnt, 48, NC);

    ln_kernel<<<ceil_div(NC, 128), 128, 0, stream>>>(s, 128, v, lp.ln1.w, lp.ln1.b,
                                                     Xf0, 160, nv, NC, 128, 16);
    run_gvp(stream, lp.ff0, NC, Xf0, 160, nv, vhf0, Xf1, 576, Gf, Vf0, true);
    run_gvp(stream, lp.ff1, NC, Xf1, 576, Vf0, vhf1, Sf, 128, Gf, Vf1, false);
    add_inplace<<<ceil_div(NC * 128, 256), 256, 0, stream>>>(s, Sf, NC * 128);
    add_inplace<<<ceil_div(NC * 48, 256), 256, 0, stream>>>(v, Vf1, NC * 48);
  }

  // 5) pool conformations, output head
  pool_conf<<<ceil_div(N_NODES * 128, 256), 256, 0, stream>>>(s, sp, 128, N_NODES);
  pool_conf<<<ceil_div(N_NODES * 48, 256), 256, 0, stream>>>(v, vp, 48, N_NODES);
  ln_kernel<<<ceil_div(N_NODES, 128), 128, 0, stream>>>(sp, 128, vp, out_ln.w, out_ln.b,
                                                        Xo1, 160, vpl, N_NODES, 128, 16);
  run_gvp(stream, out_gvp1, N_NODES, Xo1, 160, vpl, vho, Xo2, 160, Go, Vo, false);
  run_gvp(stream, out_gvp2, N_NODES, Xo2, 160, Vo, vho2, (float*)d_out, 4,
          nullptr, nullptr, false);
}
